// MoEFFNLayer_55430847922819
// MI455X (gfx1250) — compile-verified, matrix operand order checked by reading, not run on hardware
//
#include <hip/hip_runtime.h>
#include <hip/hip_bf16.h>

// ---------------- problem constants (from reference) ----------------
#define T_TOK   16384
#define H_DIM   1024
#define F_DIM   2816
#define N_EXP   8
#define TOPK    2
#define CAP     5120           // ceil(T*1.25*K/E)
#define ECAP    (N_EXP * CAP)  // 40960

// ---------------- WMMA types ----------------
typedef __attribute__((ext_vector_type(16))) __bf16 bf16x16;
typedef __attribute__((ext_vector_type(8)))  float  f32x8;

union FragU { bf16x16 v; uint4 q[2]; };

__device__ __forceinline__ f32x8 wmma_bf16(const bf16x16& a, const bf16x16& b, const f32x8& c) {
    return __builtin_amdgcn_wmma_f32_16x16x32_bf16(false, a, false, b, (short)0, c, false, false);
}

__device__ __forceinline__ f32x8 zero8() {
    f32x8 v;
    #pragma unroll
    for (int i = 0; i < 8; ++i) v[i] = 0.0f;
    return v;
}

// --------- CDNA5 async global->LDS copy (ASYNCcnt-tracked DMA path) ---------
__device__ __forceinline__ void async_copy_b128(unsigned lds_off, const void* gaddr) {
    asm volatile("global_load_async_to_lds_b128 %0, %1, off"
                 :: "v"(lds_off), "v"((unsigned long long)(uintptr_t)gaddr)
                 : "memory");
}
#define ASYNC_WAIT(imm_str) asm volatile("s_wait_asynccnt " imm_str ::: "memory")

__device__ __forceinline__ unsigned lds_addr(const void* p) {
    return (unsigned)(uintptr_t)p;   // addrspace(3) offset; HW adds wave LDS base
}

// ================= 1. Gating: logits = x@Wg, top-2 + softmax =================
__global__ __launch_bounds__(256)
void moe_gating_kernel(const float* __restrict__ x, const float* __restrict__ Wg,
                       int* __restrict__ top_idx, float* __restrict__ top_w) {
    __shared__ float part[256 * N_EXP];
    const int t = blockIdx.x, tid = threadIdx.x;
    float acc[N_EXP];
    #pragma unroll
    for (int e = 0; e < N_EXP; ++e) acc[e] = 0.0f;

    float4 xv = *(const float4*)(x + (size_t)t * H_DIM + tid * 4);
    const float xs[4] = {xv.x, xv.y, xv.z, xv.w};
    #pragma unroll
    for (int i = 0; i < 4; ++i) {
        const float* wr = Wg + (size_t)(tid * 4 + i) * N_EXP;
        float4 w0 = *(const float4*)wr;
        float4 w1 = *(const float4*)(wr + 4);
        acc[0] += xs[i] * w0.x; acc[1] += xs[i] * w0.y;
        acc[2] += xs[i] * w0.z; acc[3] += xs[i] * w0.w;
        acc[4] += xs[i] * w1.x; acc[5] += xs[i] * w1.y;
        acc[6] += xs[i] * w1.z; acc[7] += xs[i] * w1.w;
    }
    #pragma unroll
    for (int e = 0; e < N_EXP; ++e) part[tid * N_EXP + e] = acc[e];
    __syncthreads();
    for (int off = 128; off > 0; off >>= 1) {
        if (tid < off) {
            #pragma unroll
            for (int e = 0; e < N_EXP; ++e)
                part[tid * N_EXP + e] += part[(tid + off) * N_EXP + e];
        }
        __syncthreads();
    }
    if (tid == 0) {
        float v0 = -1e30f; int e0 = 0;
        #pragma unroll
        for (int e = 0; e < N_EXP; ++e)
            if (part[e] > v0) { v0 = part[e]; e0 = e; }
        float v1 = -1e30f; int e1 = 0;
        #pragma unroll
        for (int e = 0; e < N_EXP; ++e)
            if (e != e0 && part[e] > v1) { v1 = part[e]; e1 = e; }
        float p = __expf(v1 - v0);           // <= 1
        float w0 = 1.0f / (1.0f + p);
        top_idx[t * 2]     = e0;  top_idx[t * 2 + 1] = e1;
        top_w[t * 2]       = w0;  top_w[t * 2 + 1]   = 1.0f - w0;
    }
}

// ====== 2. Order-preserving dispatch scan: 3-phase, 8 waves, deterministic ======
// phase 1: each wave counts experts in its contiguous chunk (ballot popcounts)
// phase 2: exclusive prefix over chunks; phase 3: re-scan assigning positions.
__global__ __launch_bounds__(256)
void moe_scan_kernel(const int* __restrict__ top_idx, int* __restrict__ slot) {
    __shared__ int cnt[8][N_EXP];
    __shared__ int pre[8][N_EXP];
    const int tid = threadIdx.x, lane = tid & 31, w = tid >> 5;
    const unsigned ltmask = (1u << lane) - 1u;
    const int chunk = (T_TOK * TOPK) / 8;          // 4096
    const int a_begin = w * chunk, a_end = a_begin + chunk;

    int base[N_EXP];
    #pragma unroll
    for (int j = 0; j < N_EXP; ++j) base[j] = 0;

    for (int a0 = a_begin; a0 < a_end; a0 += 32) {
        const int e = top_idx[a0 + lane];
        #pragma unroll
        for (int j = 0; j < N_EXP; ++j)
            base[j] += __popc((unsigned)__ballot(e == j));
    }
    if (lane == 0) {
        #pragma unroll
        for (int j = 0; j < N_EXP; ++j) cnt[w][j] = base[j];
    }
    __syncthreads();
    if (tid == 0) {
        #pragma unroll
        for (int j = 0; j < N_EXP; ++j) {
            int run = 0;
            for (int w2 = 0; w2 < 8; ++w2) { pre[w2][j] = run; run += cnt[w2][j]; }
        }
    }
    __syncthreads();
    #pragma unroll
    for (int j = 0; j < N_EXP; ++j) base[j] = pre[w][j];

    for (int a0 = a_begin; a0 < a_end; a0 += 32) {
        const int a = a0 + lane;
        const int e = top_idx[a];
        int pos = 0;
        #pragma unroll
        for (int j = 0; j < N_EXP; ++j) {
            unsigned m = (unsigned)__ballot(e == j);
            if (e == j) pos = base[j] + __popc(m & ltmask);
            base[j] += __popc(m);
        }
        slot[a] = (pos < CAP) ? (e * CAP + pos) : ECAP;   // ECAP == dropped
    }
}

// ========== 3. Scatter x rows (f32 -> bf16) into dispatch buffer ==========
__global__ __launch_bounds__(256)
void moe_scatter_kernel(const float* __restrict__ x, const int* __restrict__ slot,
                        __bf16* __restrict__ Xbuf) {
    const int a = blockIdx.x;
    const int s = slot[a];
    if (s >= ECAP) return;                    // dropped assignment
    const int tok = a >> 1;
    const int tid = threadIdx.x;
    float4 v = *(const float4*)(x + (size_t)tok * H_DIM + tid * 4);
    union { __bf16 b[4]; uint2 u; } p;
    p.b[0] = (__bf16)v.x; p.b[1] = (__bf16)v.y; p.b[2] = (__bf16)v.z; p.b[3] = (__bf16)v.w;
    *(uint2*)(Xbuf + (size_t)s * H_DIM + tid * 4) = p.u;
}

// ===== 4. Transpose+convert weights: f32 [R,C] -> bf16 [C,R] (per expert) =====
__global__ __launch_bounds__(256)
void transpose_bf16_kernel(const float* __restrict__ src, __bf16* __restrict__ dst,
                           int R, int C) {
    __shared__ float tile[64][65];
    const int e  = blockIdx.z;
    const int c0 = blockIdx.x * 64;
    const int r0 = blockIdx.y * 64;
    const int tid = threadIdx.x;
    const float* s = src + (size_t)e * R * C;
    __bf16*      d = dst + (size_t)e * R * C;

    #pragma unroll
    for (int i = 0; i < 4; ++i) {
        int row = (tid >> 4) + i * 16;
        int col = (tid & 15) * 4;
        float4 v = *(const float4*)(s + (size_t)(r0 + row) * C + c0 + col);
        tile[row][col + 0] = v.x; tile[row][col + 1] = v.y;
        tile[row][col + 2] = v.z; tile[row][col + 3] = v.w;
    }
    __syncthreads();
    #pragma unroll
    for (int i = 0; i < 4; ++i) {
        int crow = (tid >> 4) + i * 16;
        int rc   = (tid & 15) * 4;
        union { __bf16 b[4]; uint2 u; } p;
        #pragma unroll
        for (int j = 0; j < 4; ++j) p.b[j] = (__bf16)tile[rc + j][crow];
        *(uint2*)(d + (size_t)(c0 + crow) * R + r0 + rc) = p.u;
    }
}

// =================== GEMM tiling parameters ===================
#define BM  128
#define BN  64
#define BK  64          // two WMMA k-steps per stage
#define LDK 72          // padded LDS stride (elems); 144B rows stay 16B-aligned

// ====== 5. Fused up-projection: S = silu(X@W1) * (X@W3), bf16 out ======
// 256 thr (8 waves); tile 128x64; double-buffered LDS fed by async DMA.
__global__ __launch_bounds__(256)
void moe_gemm_up_kernel(const __bf16* __restrict__ Xbuf, const __bf16* __restrict__ W1t,
                        const __bf16* __restrict__ W3t, __bf16* __restrict__ S) {
    __shared__ __bf16 As [2][BM * LDK];
    __shared__ __bf16 B1s[2][BN * LDK];
    __shared__ __bf16 B3s[2][BN * LDK];

    const int tid  = threadIdx.x;
    const int lane = tid & 31, wave = tid >> 5;
    const int wm = (wave & 3) * 32;
    const int wn = (wave >> 2) * 32;
    const int half = lane >> 4, lr = lane & 15;

    const int mBlock = blockIdx.y * BM;       // global row in [0, ECAP)
    const int nBlock = blockIdx.x * BN;       // within F
    const int e = mBlock / CAP;               // CAP % BM == 0

    const __bf16* A  = Xbuf + (size_t)mBlock * H_DIM;
    const __bf16* B1 = W1t + (size_t)e * F_DIM * H_DIM + (size_t)nBlock * H_DIM;
    const __bf16* B3 = W3t + (size_t)e * F_DIM * H_DIM + (size_t)nBlock * H_DIM;

    // per-thread copy chunk mapping (16B chunks): A 128x64 -> 4/thr, B 64x64 -> 2/thr
    int rA[4], kA[4], rB[2], kB[2];
    #pragma unroll
    for (int i = 0; i < 4; ++i) { int cid = i * 256 + tid; rA[i] = cid >> 3; kA[i] = (cid & 7) * 8; }
    #pragma unroll
    for (int i = 0; i < 2; ++i) { int cid = i * 256 + tid; rB[i] = cid >> 3; kB[i] = (cid & 7) * 8; }

    f32x8 accg[2][2], accu[2][2];
    #pragma unroll
    for (int mi = 0; mi < 2; ++mi)
        #pragma unroll
        for (int ni = 0; ni < 2; ++ni) { accg[mi][ni] = zero8(); accu[mi][ni] = zero8(); }

    auto issue = [&](int k0, int buf) {
        #pragma unroll
        for (int i = 0; i < 4; ++i)
            async_copy_b128(lds_addr(&As[buf][rA[i] * LDK + kA[i]]),
                            A + (size_t)rA[i] * H_DIM + k0 + kA[i]);
        #pragma unroll
        for (int i = 0; i < 2; ++i) {
            async_copy_b128(lds_addr(&B1s[buf][rB[i] * LDK + kB[i]]),
                            B1 + (size_t)rB[i] * H_DIM + k0 + kB[i]);
            async_copy_b128(lds_addr(&B3s[buf][rB[i] * LDK + kB[i]]),
                            B3 + (size_t)rB[i] * H_DIM + k0 + kB[i]);
        }
    };

    issue(0, 0);
    for (int k0 = 0, stage = 0; k0 < H_DIM; k0 += BK, ++stage) {
        const int cur = stage & 1, nxt = cur ^ 1;
        const bool more = (k0 + BK < H_DIM);
        __syncthreads();                       // all waves done reading buf[nxt]
        if (more) { issue(k0 + BK, nxt); ASYNC_WAIT("0x8"); }
        else      { ASYNC_WAIT("0x0"); }
        __syncthreads();                       // cur stage visible to all waves

        #pragma unroll
        for (int ks = 0; ks < 2; ++ks) {
            FragU a[2], b1[2], b3[2];
            #pragma unroll
            for (int mi = 0; mi < 2; ++mi) {
                const __bf16* p = &As[cur][(wm + mi * 16 + lr) * LDK + ks * 32 + half * 8];
                a[mi].q[0] = *(const uint4*)p;
                a[mi].q[1] = *(const uint4*)(p + 16);
            }
            #pragma unroll
            for (int ni = 0; ni < 2; ++ni) {
                const __bf16* p1 = &B1s[cur][(wn + ni * 16 + lr) * LDK + ks * 32 + half * 8];
                b1[ni].q[0] = *(const uint4*)p1;
                b1[ni].q[1] = *(const uint4*)(p1 + 16);
                const __bf16* p3 = &B3s[cur][(wn + ni * 16 + lr) * LDK + ks * 32 + half * 8];
                b3[ni].q[0] = *(const uint4*)p3;
                b3[ni].q[1] = *(const uint4*)(p3 + 16);
            }
            #pragma unroll
            for (int mi = 0; mi < 2; ++mi)
                #pragma unroll
                for (int ni = 0; ni < 2; ++ni) {
                    accg[mi][ni] = wmma_bf16(a[mi].v, b1[ni].v, accg[mi][ni]);
                    accu[mi][ni] = wmma_bf16(a[mi].v, b3[ni].v, accu[mi][ni]);
                }
        }
    }

    // SwiGLU epilogue -> S (bf16). C-layout: lanes 0-15 -> M=r, 16-31 -> M=8+r
    #pragma unroll
    for (int mi = 0; mi < 2; ++mi)
        #pragma unroll
        for (int ni = 0; ni < 2; ++ni) {
            const int col = nBlock + wn + ni * 16 + lr;
            #pragma unroll
            for (int r = 0; r < 8; ++r) {
                float g = accg[mi][ni][r];
                float u = accu[mi][ni][r];
                float sv = (g / (1.0f + __expf(-g))) * u;
                int row = mBlock + wm + mi * 16 + (half ? 8 + r : r);
                S[(size_t)row * F_DIM + col] = (__bf16)sv;
            }
        }
}

// ====== 6. Down-projection: EO = S @ W2 (f32 out) ======
__global__ __launch_bounds__(256)
void moe_gemm_down_kernel(const __bf16* __restrict__ S, const __bf16* __restrict__ W2t,
                          float* __restrict__ EO) {
    __shared__ __bf16 As[2][BM * LDK];
    __shared__ __bf16 Bs[2][BN * LDK];

    const int tid  = threadIdx.x;
    const int lane = tid & 31, wave = tid >> 5;
    const int wm = (wave & 3) * 32;
    const int wn = (wave >> 2) * 32;
    const int half = lane >> 4, lr = lane & 15;

    const int mBlock = blockIdx.y * BM;
    const int nBlock = blockIdx.x * BN;       // within H
    const int e = mBlock / CAP;

    const __bf16* A = S   + (size_t)mBlock * F_DIM;
    const __bf16* B = W2t + (size_t)e * H_DIM * F_DIM + (size_t)nBlock * F_DIM;

    int rA[4], kA[4], rB[2], kB[2];
    #pragma unroll
    for (int i = 0; i < 4; ++i) { int cid = i * 256 + tid; rA[i] = cid >> 3; kA[i] = (cid & 7) * 8; }
    #pragma unroll
    for (int i = 0; i < 2; ++i) { int cid = i * 256 + tid; rB[i] = cid >> 3; kB[i] = (cid & 7) * 8; }

    f32x8 acc[2][2];
    #pragma unroll
    for (int mi = 0; mi < 2; ++mi)
        #pragma unroll
        for (int ni = 0; ni < 2; ++ni) acc[mi][ni] = zero8();

    auto issue = [&](int k0, int buf) {
        #pragma unroll
        for (int i = 0; i < 4; ++i)
            async_copy_b128(lds_addr(&As[buf][rA[i] * LDK + kA[i]]),
                            A + (size_t)rA[i] * F_DIM + k0 + kA[i]);
        #pragma unroll
        for (int i = 0; i < 2; ++i)
            async_copy_b128(lds_addr(&Bs[buf][rB[i] * LDK + kB[i]]),
                            B + (size_t)rB[i] * F_DIM + k0 + kB[i]);
    };

    issue(0, 0);
    for (int k0 = 0, stage = 0; k0 < F_DIM; k0 += BK, ++stage) {
        const int cur = stage & 1, nxt = cur ^ 1;
        const bool more = (k0 + BK < F_DIM);
        __syncthreads();
        if (more) { issue(k0 + BK, nxt); ASYNC_WAIT("0x6"); }
        else      { ASYNC_WAIT("0x0"); }
        __syncthreads();

        #pragma unroll
        for (int ks = 0; ks < 2; ++ks) {
            FragU a[2], b[2];
            #pragma unroll
            for (int mi = 0; mi < 2; ++mi) {
                const __bf16* p = &As[cur][(wm + mi * 16 + lr) * LDK + ks * 32 + half * 8];
                a[mi].q[0] = *(const uint4*)p;
                a[mi].q[1] = *(const uint4*)(p + 16);
            }
            #pragma unroll
            for (int ni = 0; ni < 2; ++ni) {
                const __bf16* p = &Bs[cur][(wn + ni * 16 + lr) * LDK + ks * 32 + half * 8];
                b[ni].q[0] = *(const uint4*)p;
                b[ni].q[1] = *(const uint4*)(p + 16);
            }
            #pragma unroll
            for (int mi = 0; mi < 2; ++mi)
                #pragma unroll
                for (int ni = 0; ni < 2; ++ni)
                    acc[mi][ni] = wmma_bf16(a[mi].v, b[ni].v, acc[mi][ni]);
        }
    }

    #pragma unroll
    for (int mi = 0; mi < 2; ++mi)
        #pragma unroll
        for (int ni = 0; ni < 2; ++ni) {
            const int col = nBlock + wn + ni * 16 + lr;
            #pragma unroll
            for (int r = 0; r < 8; ++r) {
                int row = mBlock + wm + mi * 16 + (half ? 8 + r : r);
                EO[(size_t)row * H_DIM + col] = acc[mi][ni][r];
            }
        }
}

// ====== 7. Weighted combine: out[t] = sum_k w[t,k] * EO[slot[t,k]] ======
__global__ __launch_bounds__(256)
void moe_combine_kernel(const float* __restrict__ EO, const int* __restrict__ slot,
                        const float* __restrict__ top_w, float* __restrict__ out) {
    const int t = blockIdx.x, tid = threadIdx.x;
    int s0 = slot[t * 2], s1 = slot[t * 2 + 1];
    float w0 = (s0 < ECAP) ? top_w[t * 2]     : 0.0f;
    float w1 = (s1 < ECAP) ? top_w[t * 2 + 1] : 0.0f;
    if (s0 >= ECAP) s0 = 0;
    if (s1 >= ECAP) s1 = 0;
    const int h = tid * 4;
    float4 a = *(const float4*)(EO + (size_t)s0 * H_DIM + h);
    float4 b = *(const float4*)(EO + (size_t)s1 * H_DIM + h);
    float4 o;
    o.x = w0 * a.x + w1 * b.x;  o.y = w0 * a.y + w1 * b.y;
    o.z = w0 * a.z + w1 * b.z;  o.w = w0 * a.w + w1 * b.w;
    *(float4*)(out + (size_t)t * H_DIM + h) = o;
}

// =========================== launch ===========================
static inline size_t align256(size_t x) { return (x + 255) & ~(size_t)255; }

extern "C" void kernel_launch(void* const* d_in, const int* in_sizes, int n_in,
                              void* d_out, int out_size, void* d_ws, size_t ws_size,
                              hipStream_t stream) {
    const float* x  = (const float*)d_in[0];   // [T, H]
    const float* Wg = (const float*)d_in[1];   // [H, E]
    const float* W1 = (const float*)d_in[2];   // [E, H, F]
    const float* W3 = (const float*)d_in[3];   // [E, H, F]
    const float* W2 = (const float*)d_in[4];   // [E, F, H]
    float* out = (float*)d_out;                // [T, H]

    char* ws = (char*)d_ws;
    size_t off = 0;
    int*    top_idx = (int*)(ws + off);    off = align256(off + sizeof(int)   * T_TOK * TOPK);
    float*  top_w   = (float*)(ws + off);  off = align256(off + sizeof(float) * T_TOK * TOPK);
    int*    slot    = (int*)(ws + off);    off = align256(off + sizeof(int)   * T_TOK * TOPK);
    __bf16* Xbuf    = (__bf16*)(ws + off); off = align256(off + sizeof(__bf16) * (size_t)ECAP * H_DIM);
    __bf16* W1t     = (__bf16*)(ws + off); off = align256(off + sizeof(__bf16) * (size_t)N_EXP * F_DIM * H_DIM);
    __bf16* W3t     = (__bf16*)(ws + off); off = align256(off + sizeof(__bf16) * (size_t)N_EXP * F_DIM * H_DIM);
    __bf16* W2t     = (__bf16*)(ws + off); off = align256(off + sizeof(__bf16) * (size_t)N_EXP * H_DIM * F_DIM);
    __bf16* Sbuf    = (__bf16*)(ws + off); off = align256(off + sizeof(__bf16) * (size_t)ECAP * F_DIM);
    float*  EO      = (float*)(ws + off);  off = align256(off + sizeof(float)  * (size_t)ECAP * H_DIM);
    (void)ws_size; (void)in_sizes; (void)n_in; (void)out_size;

    moe_gating_kernel<<<T_TOK, 256, 0, stream>>>(x, Wg, top_idx, top_w);
    moe_scan_kernel<<<1, 256, 0, stream>>>(top_idx, slot);
    moe_scatter_kernel<<<T_TOK * TOPK, 256, 0, stream>>>(x, slot, Xbuf);
    transpose_bf16_kernel<<<dim3(F_DIM / 64, H_DIM / 64, N_EXP), 256, 0, stream>>>(W1, W1t, H_DIM, F_DIM);
    transpose_bf16_kernel<<<dim3(F_DIM / 64, H_DIM / 64, N_EXP), 256, 0, stream>>>(W3, W3t, H_DIM, F_DIM);
    transpose_bf16_kernel<<<dim3(H_DIM / 64, F_DIM / 64, N_EXP), 256, 0, stream>>>(W2, W2t, F_DIM, H_DIM);
    moe_gemm_up_kernel<<<dim3(F_DIM / BN, ECAP / BM), 256, 0, stream>>>(Xbuf, W1t, W3t, Sbuf);
    moe_gemm_down_kernel<<<dim3(H_DIM / BN, ECAP / BM), 256, 0, stream>>>(Sbuf, W2t, EO);
    moe_combine_kernel<<<T_TOK, 256, 0, stream>>>(EO, slot, top_w, out);
}
